// Qfnn_38543036514520
// MI455X (gfx1250) — matrix operand order repaired and checked by reading, not compiled, not run on hardware
//
#include <hip/hip_runtime.h>

// MI455X / gfx1250 fused forward pass.
// conv2/conv3 via implicit-GEMM f16 WMMA (16x16x32, f32 accum), NHWC padded
// activations, fragment-packed weights, fused BN+ReLU+maxpool epilogue.

typedef __attribute__((ext_vector_type(16))) _Float16 v16h;
typedef __attribute__((ext_vector_type(8)))  float    v8f;

#define BN_INV 0.9999950000374997f   // 1/sqrt(1 + 1e-5)

union Frag16 { v16h v; uint4 u[2]; };

static __device__ __forceinline__ v8f wmma16(v16h a, v16h b, v8f c) {
  return __builtin_amdgcn_wmma_f32_16x16x32_f16(false, a, false, b, (short)0, c, false, false);
}

// ---------------------------------------------------------------------------
// Weight fragment packing: B-fragment layout (wave32):
//   lane 0-15 : K = 0..15  (K = j),  lane 16-31 : K = 16..31 (K = 16+j)
//   column N  = lane % 16
// packed[((s*NT + nt)*32 + lane)*16 + j] = W[N = nt*16+lane%16][K = s*32+16*(lane/16)+j]
// where global K -> (pos = K/IC, ic = K%IC, ky = pos/3, kx = pos%3), OIHW weights.
// ---------------------------------------------------------------------------
__global__ void pack_weights(const float* __restrict__ c2w, const float* __restrict__ c3w,
                             _Float16* __restrict__ w2pk, _Float16* __restrict__ w3pk) {
  const int N2 = 9 * 4 * 32 * 16;     // conv2: 9 ksteps, NT=4
  const int N3 = 18 * 8 * 32 * 16;    // conv3: 18 ksteps, NT=8
  int t = blockIdx.x * 256 + threadIdx.x;
  if (t < N2) {
    int j = t & 15; int q = t >> 4;
    int lane = q & 31; q >>= 5;
    int nt = q & 3; int s = q >> 2;                 // 0..8
    int K = s * 32 + ((lane >> 4) << 4) + j;        // 0..287
    int oc = nt * 16 + (lane & 15);
    int ic = K & 31; int pos = K >> 5;              // IC = 32
    int ky = pos / 3, kx = pos - 3 * (pos / 3);
    w2pk[t] = (_Float16)c2w[((oc * 32 + ic) * 3 + ky) * 3 + kx];
  } else if (t < N2 + N3) {
    int u = t - N2;
    int j = u & 15; int q = u >> 4;
    int lane = q & 31; q >>= 5;
    int nt = q & 7; int s = q >> 3;                 // 0..17
    int K = s * 32 + ((lane >> 4) << 4) + j;        // 0..575
    int oc = nt * 16 + (lane & 15);
    int ic = K & 63; int pos = K >> 6;              // IC = 64
    int ky = pos / 3, kx = pos - 3 * (pos / 3);
    w3pk[u] = (_Float16)c3w[((oc * 64 + ic) * 3 + ky) * 3 + kx];
  }
}

// ---------------------------------------------------------------------------
// conv1 (IC=1) + BN + ReLU + 2x2 pool, direct. Output padded NHWC f16
// [B,16,16,32] with zero border (interior pixel (y,x) -> (y+1, x+1)).
// ---------------------------------------------------------------------------
__global__ void conv1_bn_relu_pool(const float* __restrict__ x,
                                   const float* __restrict__ w,
                                   const float* __restrict__ cb,
                                   const float* __restrict__ g,
                                   const float* __restrict__ bb,
                                   _Float16* __restrict__ out, int B) {
  int t = blockIdx.x * 256 + threadIdx.x;
  if (t >= B * 8192) return;                 // 16*16*32 = 8192 per image
  int c = t & 31; int rest = t >> 5;
  int xp = rest & 15; rest >>= 4;
  int yp = rest & 15; int b = rest >> 4;
  if (yp == 0 || yp == 15 || xp == 0 || xp == 15) { out[t] = (_Float16)0.f; return; }
  int y = yp - 1, x0 = xp - 1;               // pooled output coords 0..13
  float wv[9];
#pragma unroll
  for (int i = 0; i < 9; ++i) wv[i] = w[c * 9 + i];
  float scale = g[c] * BN_INV;
  float bias  = cb[c] * scale + bb[c];
  const float* xb = x + (size_t)b * 784;
  float best = -3.4e38f;
#pragma unroll
  for (int dy = 0; dy < 2; ++dy)
#pragma unroll
    for (int dx = 0; dx < 2; ++dx) {
      int oy = 2 * y + dy, ox = 2 * x0 + dx;
      float acc = 0.f;
#pragma unroll
      for (int ky = 0; ky < 3; ++ky) {
        int iy = oy + ky - 1;
        if (iy < 0 || iy >= 28) continue;
#pragma unroll
        for (int kx = 0; kx < 3; ++kx) {
          int ix = ox + kx - 1;
          if (ix < 0 || ix >= 28) continue;
          acc += xb[iy * 28 + ix] * wv[ky * 3 + kx];
        }
      }
      best = fmaxf(best, acc * scale + bias);
    }
  out[t] = (_Float16)fmaxf(best, 0.f);
}

__global__ void zero_u4(uint4* __restrict__ p, int n) {
  int t = blockIdx.x * 256 + threadIdx.x;
  if (t < n) { uint4 z; z.x = z.y = z.z = z.w = 0u; p[t] = z; }
}

// ---------------------------------------------------------------------------
// conv2: implicit GEMM, M-tile = 2 rows x 8 cols of 14x14 output, N=64 (NT=4),
// K = 9*32 = 288 (9 WMMA K-steps). Fused BN+ReLU+pool -> padded [B,9,9,64] f16.
// ---------------------------------------------------------------------------
__global__ void conv2_wmma_pool(const _Float16* __restrict__ act,
                                const _Float16* __restrict__ wpk,
                                const float* __restrict__ cb,
                                const float* __restrict__ g,
                                const float* __restrict__ bb,
                                _Float16* __restrict__ outp, int nTiles) {
  int lane = threadIdx.x & 31;
  int tile = blockIdx.x * 8 + (threadIdx.x >> 5);
  if (tile >= nTiles) return;                      // wave-uniform: EXEC stays full
  int b = tile / 14; int rr = tile - b * 14;
  int ty = rr >> 1, tx = rr & 1;
  int laneHi = lane >> 4, mrow = lane & 15;
  int y = 2 * ty + (mrow >> 3);                    // 0..13
  int xx = 8 * tx + (mrow & 7); if (xx > 13) xx = 13;  // clamp (garbage cols dropped)
  v8f acc[4];
#pragma unroll
  for (int nt = 0; nt < 4; ++nt) { v8f z = {0.f,0.f,0.f,0.f,0.f,0.f,0.f,0.f}; acc[nt] = z; }
  for (int s = 0; s < 9; ++s) {
    int ky = s / 3, kx = s - 3 * (s / 3);
    // A fragment: lane half picks channel chunk {c0..c0+15}, c0 = 16*laneHi
    const _Float16* ap = act + ((((size_t)b * 16 + (y + ky)) * 16 + (xx + kx)) * 32 + (laneHi << 4));
    Frag16 A_; A_.u[0] = *(const uint4*)ap; A_.u[1] = *(const uint4*)(ap + 8);
#pragma unroll
    for (int nt = 0; nt < 4; ++nt) {
      const _Float16* bp = wpk + (((size_t)(s * 4 + nt) * 32 + lane) << 4);
      Frag16 B_; B_.u[0] = *(const uint4*)bp; B_.u[1] = *(const uint4*)(bp + 8);
      acc[nt] = wmma16(A_.v, B_.v, acc[nt]);
    }
  }
  // Epilogue: C/D layout -> lane half = row (M 0-7 / 8-15), VGPR index = column.
  int ncol = lane & 15;
#pragma unroll
  for (int nt = 0; nt < 4; ++nt) {
    int ch = nt * 16 + ncol;
    float scale = g[ch] * BN_INV;
    float bias  = cb[ch] * scale + bb[ch];
#pragma unroll
    for (int p = 0; p < 4; ++p) {
      float v0 = acc[nt][2 * p] * scale + bias;
      float v1 = acc[nt][2 * p + 1] * scale + bias;
      float mx = fmaxf(v0, v1);                      // horizontal pool
      float other = __shfl_xor(mx, 16, 32);          // vertical pool (row swap)
      mx = fmaxf(fmaxf(mx, other), 0.f);             // + ReLU
      int px = 4 * tx + p;
      if (laneHi == 0 && px < 7)
        outp[(((size_t)b * 9 + 1 + ty) * 9 + 1 + px) * 64 + ch] = (_Float16)mx;
    }
  }
}

// ---------------------------------------------------------------------------
// conv3: N=128 (NT=8), K = 9*64 = 576 (18 K-steps), pool 7x7 -> 3x3.
// Output [B,3,3,128] f32 (unpadded).
// ---------------------------------------------------------------------------
__global__ void conv3_wmma_pool(const _Float16* __restrict__ act,
                                const _Float16* __restrict__ wpk,
                                const float* __restrict__ cb,
                                const float* __restrict__ g,
                                const float* __restrict__ bb,
                                float* __restrict__ outp, int nTiles) {
  int lane = threadIdx.x & 31;
  int tile = blockIdx.x * 8 + (threadIdx.x >> 5);
  if (tile >= nTiles) return;
  int b = tile / 3; int ty = tile - b * 3;
  int laneHi = lane >> 4, mrow = lane & 15;
  int y = 2 * ty + (mrow >> 3);                    // 0..5 (row 6 unused by pool)
  int xx = (mrow & 7); if (xx > 6) xx = 6;         // col 7 garbage, dropped
  v8f acc[8];
#pragma unroll
  for (int nt = 0; nt < 8; ++nt) { v8f z = {0.f,0.f,0.f,0.f,0.f,0.f,0.f,0.f}; acc[nt] = z; }
  for (int s = 0; s < 18; ++s) {
    int pos = s >> 1;
    int cOff = (s & 1) << 5;
    int ky = pos / 3, kx = pos - 3 * (pos / 3);
    const _Float16* ap = act + ((((size_t)b * 9 + (y + ky)) * 9 + (xx + kx)) * 64 + cOff + (laneHi << 4));
    Frag16 A_; A_.u[0] = *(const uint4*)ap; A_.u[1] = *(const uint4*)(ap + 8);
#pragma unroll
    for (int nt = 0; nt < 8; ++nt) {
      const _Float16* bp = wpk + (((size_t)(s * 8 + nt) * 32 + lane) << 4);
      Frag16 B_; B_.u[0] = *(const uint4*)bp; B_.u[1] = *(const uint4*)(bp + 8);
      acc[nt] = wmma16(A_.v, B_.v, acc[nt]);
    }
  }
  int ncol = lane & 15;
#pragma unroll
  for (int nt = 0; nt < 8; ++nt) {
    int ch = nt * 16 + ncol;
    float scale = g[ch] * BN_INV;
    float bias  = cb[ch] * scale + bb[ch];
#pragma unroll
    for (int p = 0; p < 4; ++p) {
      float v0 = acc[nt][2 * p] * scale + bias;
      float v1 = acc[nt][2 * p + 1] * scale + bias;
      float mx = fmaxf(v0, v1);
      float other = __shfl_xor(mx, 16, 32);
      mx = fmaxf(fmaxf(mx, other), 0.f);
      if (laneHi == 0 && p < 3)
        outp[(((size_t)b * 3 + ty) * 3 + p) * 128 + ch] = mx;
    }
  }
}

// ---------------------------------------------------------------------------
// Head: linear 1152->3, GroupNorm(1,3), fuzzy membership, 3-qubit circuit,
// tiled MLP 81->128->64->10 + softmax. One thread per batch element.
// ---------------------------------------------------------------------------
struct c2x { float re, im; };
static __device__ __forceinline__ c2x cmul(c2x a, c2x b) {
  c2x r; r.re = a.re * b.re - a.im * b.im; r.im = a.re * b.im + a.im * b.re; return r;
}
static __device__ __forceinline__ c2x cadd(c2x a, c2x b) {
  c2x r; r.re = a.re + b.re; r.im = a.im + b.im; return r;
}
static __device__ void build_u(float a, float mAng, float bAng, c2x U[2][2]) {
  float ca = cosf(0.5f * a),    sa = sinf(0.5f * a);
  float cm = cosf(0.5f * mAng), sm = sinf(0.5f * mAng);
  float cb = cosf(0.5f * bAng), sb = sinf(0.5f * bAng);
  c2x RXa[2][2] = {{{ca, 0.f}, {0.f, -sa}}, {{0.f, -sa}, {ca, 0.f}}};
  c2x RZm[2][2] = {{{cm, -sm}, {0.f, 0.f}}, {{0.f, 0.f}, {cm, sm}}};
  c2x RXb[2][2] = {{{cb, 0.f}, {0.f, -sb}}, {{0.f, -sb}, {cb, 0.f}}};
  c2x T[2][2];
#pragma unroll
  for (int r = 0; r < 2; ++r)
#pragma unroll
    for (int c = 0; c < 2; ++c)
      T[r][c] = cadd(cmul(RZm[r][0], RXa[0][c]), cmul(RZm[r][1], RXa[1][c]));
#pragma unroll
  for (int r = 0; r < 2; ++r)
#pragma unroll
    for (int c = 0; c < 2; ++c)
      U[r][c] = cadd(cmul(RXb[r][0], T[0][c]), cmul(RXb[r][1], T[1][c]));
}

__global__ void head_kernel(const float* __restrict__ p3, int B,
                            const float* __restrict__ lw, const float* __restrict__ lb,
                            const float* __restrict__ gg, const float* __restrict__ gb,
                            const float* __restrict__ fm, const float* __restrict__ th,
                            const float* __restrict__ qw,
                            const float* __restrict__ h1w, const float* __restrict__ h1b,
                            const float* __restrict__ g1,  const float* __restrict__ b1,
                            const float* __restrict__ h2w, const float* __restrict__ h2b,
                            const float* __restrict__ g2,  const float* __restrict__ b2,
                            const float* __restrict__ cw,  const float* __restrict__ cbv,
                            float* __restrict__ out) {
  __shared__ float h1s[64 * 129];
  __shared__ float h2s[64 * 65];
  int tid = threadIdx.x;
  int b = blockIdx.x * 64 + tid;
  if (b >= B) return;

  // linear 1152->3 ; flatten order is NCHW: feature = c*9 + y*3 + x
  float a0 = lb[0], a1 = lb[1], a2 = lb[2];
  for (int y = 0; y < 3; ++y)
    for (int x = 0; x < 3; ++x) {
      const float* pp = p3 + (((size_t)b * 3 + y) * 3 + x) * 128;
      int fo = y * 3 + x;
      for (int c = 0; c < 128; ++c) {
        float v = pp[c];
        int f = c * 9 + fo;
        a0 += v * lw[f]; a1 += v * lw[1152 + f]; a2 += v * lw[2304 + f];
      }
    }
  float xs[3] = {a0, a1, a2};
  float mu = (xs[0] + xs[1] + xs[2]) * (1.f / 3.f);
  float var = ((xs[0] - mu) * (xs[0] - mu) + (xs[1] - mu) * (xs[1] - mu) +
               (xs[2] - mu) * (xs[2] - mu)) * (1.f / 3.f);
  float inv = rsqrtf(var + 1e-5f);
#pragma unroll
  for (int i = 0; i < 3; ++i) xs[i] = (xs[i] - mu) * inv * gg[i] + gb[i];

  // fuzzy membership -> angles
  float ang[3];
#pragma unroll
  for (int i = 0; i < 3; ++i) {
    float fs = 0.f;
#pragma unroll
    for (int j = 0; j < 3; ++j) {
      float d = (xs[i] - fm[j * 3 + i]) / th[j * 3 + i];
      fs += expf(-0.5f * d * d);
    }
    float fz = fs * (1.f / 3.f);
    float sq = sqrtf(fz + 1e-16f);
    if (sq > 0.99999f) sq = 0.99999f;
    ang[i] = 2.f * asinf(sq);
  }

  // 3-qubit circuit: RY embedding -> CNOT01,12,20 -> RX RZ RX per wire -> CNOT01,12
  float qc[3], qs[3];
#pragma unroll
  for (int i = 0; i < 3; ++i) { qc[i] = cosf(0.5f * ang[i]); qs[i] = sinf(0.5f * ang[i]); }
  c2x psi[2][2][2];
#pragma unroll
  for (int i = 0; i < 2; ++i)
#pragma unroll
    for (int j = 0; j < 2; ++j)
#pragma unroll
      for (int k = 0; k < 2; ++k) {
        psi[i][j][k].re = (i ? qs[0] : qc[0]) * (j ? qs[1] : qc[1]) * (k ? qs[2] : qc[2]);
        psi[i][j][k].im = 0.f;
      }
#pragma unroll
  for (int k = 0; k < 2; ++k) { c2x t = psi[1][0][k]; psi[1][0][k] = psi[1][1][k]; psi[1][1][k] = t; } // CNOT01
#pragma unroll
  for (int i = 0; i < 2; ++i) { c2x t = psi[i][1][0]; psi[i][1][0] = psi[i][1][1]; psi[i][1][1] = t; } // CNOT12
#pragma unroll
  for (int j = 0; j < 2; ++j) { c2x t = psi[0][j][1]; psi[0][j][1] = psi[1][j][1]; psi[1][j][1] = t; } // CNOT20

  c2x U0[2][2], U1[2][2], U2[2][2];
  build_u(qw[0], qw[3], qw[6], U0);
  build_u(qw[1], qw[4], qw[7], U1);
  build_u(qw[2], qw[5], qw[8], U2);
#pragma unroll
  for (int j = 0; j < 2; ++j)
#pragma unroll
    for (int k = 0; k < 2; ++k) {
      c2x t0 = psi[0][j][k], t1 = psi[1][j][k];
      psi[0][j][k] = cadd(cmul(U0[0][0], t0), cmul(U0[0][1], t1));
      psi[1][j][k] = cadd(cmul(U0[1][0], t0), cmul(U0[1][1], t1));
    }
#pragma unroll
  for (int i = 0; i < 2; ++i)
#pragma unroll
    for (int k = 0; k < 2; ++k) {
      c2x t0 = psi[i][0][k], t1 = psi[i][1][k];
      psi[i][0][k] = cadd(cmul(U1[0][0], t0), cmul(U1[0][1], t1));
      psi[i][1][k] = cadd(cmul(U1[1][0], t0), cmul(U1[1][1], t1));
    }
#pragma unroll
  for (int i = 0; i < 2; ++i)
#pragma unroll
    for (int j = 0; j < 2; ++j) {
      c2x t0 = psi[i][j][0], t1 = psi[i][j][1];
      psi[i][j][0] = cadd(cmul(U2[0][0], t0), cmul(U2[0][1], t1));
      psi[i][j][1] = cadd(cmul(U2[1][0], t0), cmul(U2[1][1], t1));
    }
#pragma unroll
  for (int k = 0; k < 2; ++k) { c2x t = psi[1][0][k]; psi[1][0][k] = psi[1][1][k]; psi[1][1][k] = t; } // CNOT01
#pragma unroll
  for (int i = 0; i < 2; ++i) { c2x t = psi[i][1][0]; psi[i][1][0] = psi[i][1][1]; psi[i][1][1] = t; } // CNOT12

  float pr[2][2][2];
#pragma unroll
  for (int i = 0; i < 2; ++i)
#pragma unroll
    for (int j = 0; j < 2; ++j)
#pragma unroll
      for (int k = 0; k < 2; ++k)
        pr[i][j][k] = psi[i][j][k].re * psi[i][j][k].re + psi[i][j][k].im * psi[i][j][k].im;
  float zf[3];
  zf[0] = (pr[0][0][0] + pr[0][0][1] + pr[0][1][0] + pr[0][1][1]) -
          (pr[1][0][0] + pr[1][0][1] + pr[1][1][0] + pr[1][1][1]);
  zf[1] = (pr[0][0][0] + pr[0][0][1] + pr[1][0][0] + pr[1][0][1]) -
          (pr[0][1][0] + pr[0][1][1] + pr[1][1][0] + pr[1][1][1]);
  zf[2] = (pr[0][0][0] + pr[0][1][0] + pr[1][0][0] + pr[1][1][0]) -
          (pr[0][0][1] + pr[0][1][1] + pr[1][0][1] + pr[1][1][1]);

  // MLP: q = tile(z, 27) -> 81 ; h1 = relu(BN(q @ h1w^T + h1b))
  float* myh1 = h1s + tid * 129;
  for (int i = 0; i < 128; ++i) {
    float acc = h1b[i];
    const float* wr = h1w + i * 81;
    for (int rbl = 0; rbl < 27; ++rbl)
      acc += zf[0] * wr[rbl * 3] + zf[1] * wr[rbl * 3 + 1] + zf[2] * wr[rbl * 3 + 2];
    myh1[i] = fmaxf(acc * (g1[i] * BN_INV) + b1[i], 0.f);
  }
  float* myh2 = h2s + tid * 65;
  for (int i = 0; i < 64; ++i) {
    float acc = h2b[i];
    const float* wr = h2w + i * 128;
    for (int j = 0; j < 128; ++j) acc += myh1[j] * wr[j];
    myh2[i] = fmaxf(acc * (g2[i] * BN_INV) + b2[i], 0.f);
  }
  float lg[10]; float mxl = -3.4e38f;
#pragma unroll
  for (int i = 0; i < 10; ++i) {
    float acc = cbv[i];
    const float* wr = cw + i * 64;
    for (int j = 0; j < 64; ++j) acc += myh2[j] * wr[j];
    lg[i] = acc; mxl = fmaxf(mxl, acc);
  }
  float se = 0.f;
#pragma unroll
  for (int i = 0; i < 10; ++i) { lg[i] = expf(lg[i] - mxl); se += lg[i]; }
  float rse = 1.f / se;
#pragma unroll
  for (int i = 0; i < 10; ++i) out[(size_t)b * 10 + i] = lg[i] * rse;
}

// ---------------------------------------------------------------------------
extern "C" void kernel_launch(void* const* d_in, const int* in_sizes, int n_in,
                              void* d_out, int out_size, void* d_ws, size_t ws_size,
                              hipStream_t stream) {
  (void)n_in; (void)out_size; (void)ws_size;
  const float* X    = (const float*)d_in[0];
  const float* C1W  = (const float*)d_in[1];
  const float* C1B  = (const float*)d_in[2];
  const float* BN1G = (const float*)d_in[3];
  const float* BN1B = (const float*)d_in[4];
  const float* C2W  = (const float*)d_in[5];
  const float* C2B  = (const float*)d_in[6];
  const float* BN2G = (const float*)d_in[7];
  const float* BN2B = (const float*)d_in[8];
  const float* C3W  = (const float*)d_in[9];
  const float* C3B  = (const float*)d_in[10];
  const float* BN3G = (const float*)d_in[11];
  const float* BN3B = (const float*)d_in[12];
  const float* LINW = (const float*)d_in[13];
  const float* LINB = (const float*)d_in[14];
  const float* GNG  = (const float*)d_in[15];
  const float* GNB  = (const float*)d_in[16];
  const float* FM   = (const float*)d_in[17];
  const float* TH   = (const float*)d_in[18];
  const float* QW   = (const float*)d_in[19];
  const float* H1W  = (const float*)d_in[20];
  const float* H1B  = (const float*)d_in[21];
  const float* BH1G = (const float*)d_in[22];
  const float* BH1B = (const float*)d_in[23];
  const float* H2W  = (const float*)d_in[24];
  const float* H2B  = (const float*)d_in[25];
  const float* BH2G = (const float*)d_in[26];
  const float* BH2B = (const float*)d_in[27];
  const float* CLSW = (const float*)d_in[28];
  const float* CLSB = (const float*)d_in[29];
  const int B = in_sizes[0] / 784;          // 4096

  char* ws = (char*)d_ws;
  _Float16* w2pk = (_Float16*)(ws);                    // 36,864 B
  _Float16* w3pk = (_Float16*)(ws + 65536);            // 147,456 B
  _Float16* out1 = (_Float16*)(ws + 262144);           // [B,16,16,32] f16 (padded)
  size_t out1B = (size_t)B * 8192 * 2;
  _Float16* p2   = (_Float16*)(ws + 262144 + out1B);   // [B,9,9,64] f16 (padded)
  size_t p2B = (size_t)B * 81 * 64 * 2;
  float*    p3   = (float*)(ws + 262144 + out1B + p2B);// [B,3,3,128] f32
  float* out = (float*)d_out;

  pack_weights<<<(9 * 4 * 32 * 16 + 18 * 8 * 32 * 16 + 255) / 256, 256, 0, stream>>>(
      C2W, C3W, w2pk, w3pk);

  {
    long long n = (long long)B * 8192;
    conv1_bn_relu_pool<<<(unsigned)((n + 255) / 256), 256, 0, stream>>>(
        X, C1W, C1B, BN1G, BN1B, out1, B);
  }
  {
    int nu4 = B * 648;                                  // B*81*64*2 bytes / 16
    zero_u4<<<(nu4 + 255) / 256, 256, 0, stream>>>((uint4*)p2, nu4);
  }
  {
    int nT = B * 14;                                    // 2x8-pixel M-tiles per image
    conv2_wmma_pool<<<(nT + 7) / 8, 256, 0, stream>>>(
        out1, w2pk, C2B, BN2G, BN2B, p2, nT);
  }
  {
    int nT = B * 3;
    conv3_wmma_pool<<<(nT + 7) / 8, 256, 0, stream>>>(
        p2, w3pk, C3B, BN3G, BN3B, p3, nT);
  }
  head_kernel<<<(B + 63) / 64, 64, 0, stream>>>(
      p3, B, LINW, LINB, GNG, GNB, FM, TH, QW,
      H1W, H1B, BH1G, BH1B, H2W, H2B, BH2G, BH2B, CLSW, CLSB, out);
}
